// CANDY_82179904242298
// MI455X (gfx1250) — matrix operand order repaired and checked by reading, not compiled, not run on hardware
//
#include <hip/hip_runtime.h>
#include <hip/hip_bf16.h>

// MI455X / gfx1250 fused pipeline:
//   x = input/5000;  Xs = x * hardtanh(p_mask)                        (bf16)
//   tp = ht( ht(Wp_eff @ Xs) @ p_lin_w^T + p_lin_b )   per (b,c)      (bf16)
//   tz = ht( ht(Wzp   @ tp) @ z_lin_w^T + z_lin_b )    per (b,c)      (bf16, reuses Xs buffer)
//   comb = sum_c tp + sum_c tz + x[:,0]                               (bf16)
//   out  = relu(comb @ fc1^T + b1) @ fc2^T + b2                       (f32)
// All GEMMs: bf16 inputs, f32 accumulate via v_wmma_f32_16x16x32_bf16.
// Epilogue behavior (acts/bias/output dtype) is templated so it lowers to
// straight-line VALU instead of per-element scalar branches.

typedef __bf16 bf16;
typedef __attribute__((ext_vector_type(16))) __bf16 v16bf;
typedef __attribute__((ext_vector_type(8)))  float  v8f;

#define TILE_M    64
#define KC        32
#define N_FULL    512
#define AS_STRIDE 40    // 80B rows: 16B-aligned, conflict-free A-frag reads
#define BT_STRIDE 34    // 68B cols: conflict-free packed-pair B-frag reads
#define U_STRIDE  520   // 1040B rows: 16B-aligned, conflict-free

__device__ __forceinline__ float hardtanh_f(float x) {
    return fminf(fmaxf(x, -1.0f), 1.0f);
}

template <int MODE>
__device__ __forceinline__ float apply_act(float x) {
    if (MODE == 1) return fminf(fmaxf(x, -1.0f), 1.0f);  // hardtanh
    if (MODE == 2) return fmaxf(x, 0.0f);                // relu
    return x;
}

// A-fragment (16x32 bf16, ISA 7.12.2): lane<16 -> M=lane, K {0..7,16..23};
// lane>=16 -> M=lane-16, K {8..15,24..31}. Reads packed pairs as u32 from LDS.
__device__ __forceinline__ v16bf load_afrag(const bf16* base, int stride, int m0,
                                            int k0, int lane) {
    union { v16bf v; unsigned u[8]; } r;
    const bf16* row = base + (m0 + (lane & 15)) * stride;
    const int kb = k0 + ((lane >> 4) << 3);
#pragma unroll
    for (int v = 0; v < 8; ++v) {
        int k = kb + ((v & 3) << 1) + ((v >> 2) << 4);
        r.u[v] = *(const unsigned*)(row + k);
    }
    return r.v;
}

// B-fragment (32x16 bf16): lane<16 -> N=lane, K {0..7,16..23}; lane>=16 -> K {8..15,24..31}.
// BsT stores B transposed: BsT[n][k] = B[k][n], so pairs are contiguous.
__device__ __forceinline__ v16bf load_bfrag(const bf16* bT, int n0, int lane) {
    union { v16bf v; unsigned u[8]; } r;
    const bf16* col = bT + (n0 + (lane & 15)) * BT_STRIDE + ((lane >> 4) << 3);
#pragma unroll
    for (int v = 0; v < 8; ++v) r.u[v] = *(const unsigned*)(col + ((v & 3) << 1) + ((v >> 2) << 4));
    return r.v;
}

// One workgroup: 64 output rows x 512 cols of  out = act2(act1(A@B1+bias1)@B2+bias2).
// 256 threads = 8 wave32; each wave owns a 64-col slab (4x4 WMMA tiles).
template <int ACT1, int ACT2, bool HAS_B1, bool OUT_BF16>
__global__ __launch_bounds__(256)
void fused_dual_gemm(const bf16* __restrict__ A,  long A_bs,
                     const bf16* __restrict__ B1, long B1_bs,
                     const bf16* __restrict__ B2,
                     const float* __restrict__ bias1,
                     const float* __restrict__ bias2,
                     void* __restrict__ outp, long out_bs) {
    __shared__ bf16 As [TILE_M * AS_STRIDE];
    __shared__ bf16 BsT[N_FULL * BT_STRIDE];
    __shared__ bf16 U  [TILE_M * U_STRIDE];

    const int tid  = threadIdx.x;
    const int lane = tid & 31;
    const int w    = tid >> 5;
    const long batch = blockIdx.y;
    const long mbase = (long)blockIdx.x * TILE_M;

    const bf16* Ag  = A  + A_bs  * batch + mbase * 512;
    const bf16* B1g = B1 + B1_bs * batch;

    v8f acc[4][4];
#pragma unroll
    for (int i = 0; i < 4; ++i)
#pragma unroll
        for (int j = 0; j < 4; ++j)
#pragma unroll
            for (int e = 0; e < 8; ++e) acc[i][j][e] = 0.0f;

    // ---------------- GEMM1: T = A @ B1 ----------------
    for (int k0 = 0; k0 < 512; k0 += KC) {
        if (k0 + KC < 512)  // global_prefetch_b8 next K-chunk of B1
            __builtin_prefetch(B1g + (long)(k0 + KC) * 512 + (tid << 6), 0, 0);
        // stage A tile [64 x 32] (row-major, packed u32 pairs)
        for (int p = tid; p < TILE_M * (KC / 2); p += 256) {
            int row = p >> 4, kp = p & 15;
            unsigned val = *(const unsigned*)(Ag + (long)row * 512 + k0 + (kp << 1));
            *(unsigned*)(As + row * AS_STRIDE + (kp << 1)) = val;
        }
        // stage B chunk [32 x 512] transposed -> BsT[n][k] with packed K-pairs
        for (int p = tid; p < N_FULL * (KC / 2); p += 256) {
            int n = p & 511, kp = p >> 9;
            const bf16* src = B1g + (long)(k0 + (kp << 1)) * 512 + n;
            union { unsigned u; bf16 h[2]; } pk;
            pk.h[0] = src[0]; pk.h[1] = src[512];
            *(unsigned*)(BsT + n * BT_STRIDE + (kp << 1)) = pk.u;
        }
        __syncthreads();
        v16bf a[4], b[4];
#pragma unroll
        for (int mt = 0; mt < 4; ++mt) a[mt] = load_afrag(As, AS_STRIDE, mt << 4, 0, lane);
#pragma unroll
        for (int nt = 0; nt < 4; ++nt) b[nt] = load_bfrag(BsT, (w << 6) + (nt << 4), lane);
#pragma unroll
        for (int mt = 0; mt < 4; ++mt)
#pragma unroll
            for (int nt = 0; nt < 4; ++nt)
                acc[mt][nt] = __builtin_amdgcn_wmma_f32_16x16x32_bf16(
                    false, a[mt], false, b[nt], (short)0, acc[mt][nt], false, false);
        __syncthreads();
    }

    // epilogue 1: bias1 + act1 -> U (bf16 in LDS), reset accumulators
#pragma unroll
    for (int nt = 0; nt < 4; ++nt) {
        const int col = (w << 6) + (nt << 4) + (lane & 15);
        const float bv = HAS_B1 ? bias1[col] : 0.0f;
#pragma unroll
        for (int mt = 0; mt < 4; ++mt)
#pragma unroll
            for (int r = 0; r < 8; ++r) {
                int row = (mt << 4) + r + ((lane >> 4) << 3);
                U[row * U_STRIDE + col] = (bf16)apply_act<ACT1>(acc[mt][nt][r] + bv);
                acc[mt][nt][r] = 0.0f;
            }
    }

    // ---------------- GEMM2: D = U @ B2 ----------------
    for (int k0 = 0; k0 < 512; k0 += KC) {
        for (int p = tid; p < N_FULL * (KC / 2); p += 256) {
            int n = p & 511, kp = p >> 9;
            const bf16* src = B2 + (long)(k0 + (kp << 1)) * 512 + n;
            union { unsigned u; bf16 h[2]; } pk;
            pk.h[0] = src[0]; pk.h[1] = src[512];
            *(unsigned*)(BsT + n * BT_STRIDE + (kp << 1)) = pk.u;
        }
        __syncthreads();   // also orders epilogue-1 U writes vs U reads
        v16bf a[4], b[4];
#pragma unroll
        for (int mt = 0; mt < 4; ++mt) a[mt] = load_afrag(U, U_STRIDE, mt << 4, k0, lane);
#pragma unroll
        for (int nt = 0; nt < 4; ++nt) b[nt] = load_bfrag(BsT, (w << 6) + (nt << 4), lane);
#pragma unroll
        for (int mt = 0; mt < 4; ++mt)
#pragma unroll
            for (int nt = 0; nt < 4; ++nt)
                acc[mt][nt] = __builtin_amdgcn_wmma_f32_16x16x32_bf16(
                    false, a[mt], false, b[nt], (short)0, acc[mt][nt], false, false);
        __syncthreads();
    }

    // epilogue 2: bias2 + act2 -> global (bf16 intermediate or f32 final)
    {
        bf16*  Ob = OUT_BF16 ? ((bf16*)outp)  + out_bs * batch + mbase * 512 : nullptr;
        float* Of = OUT_BF16 ? nullptr : ((float*)outp) + out_bs * batch + mbase * 512;
#pragma unroll
        for (int nt = 0; nt < 4; ++nt) {
            const int col = (w << 6) + (nt << 4) + (lane & 15);
            const float bv = bias2[col];
#pragma unroll
            for (int mt = 0; mt < 4; ++mt)
#pragma unroll
                for (int r = 0; r < 8; ++r) {
                    int row = (mt << 4) + r + ((lane >> 4) << 3);
                    float xv = apply_act<ACT2>(acc[mt][nt][r] + bv);
                    if (OUT_BF16) Ob[(long)row * 512 + col] = (bf16)xv;
                    else          Of[(long)row * 512 + col] = xv;
                }
        }
    }
}

// Xs[b,c,h,i] = bf16( input/5000 * hardtanh(p_mask[h,i]) ), float4-vectorized.
__global__ void prep_x(const float* __restrict__ inp, const float* __restrict__ pmask,
                       bf16* __restrict__ Xs) {
    long i4 = (long)blockIdx.x * blockDim.x + threadIdx.x;   // 16777216 float4s
    const float4 xv = ((const float4*)inp)[i4];
    const float4 mv = ((const float4*)pmask)[(int)(i4 & 65535)];
    const float s = 1.0f / 5000.0f;
    bf16* dst = Xs + (i4 << 2);
    dst[0] = (bf16)(xv.x * s * hardtanh_f(mv.x));
    dst[1] = (bf16)(xv.y * s * hardtanh_f(mv.y));
    dst[2] = (bf16)(xv.z * s * hardtanh_f(mv.z));
    dst[3] = (bf16)(xv.w * s * hardtanh_f(mv.w));
}

// bf16 weight prep: Wp_eff = Wp + diag(Wp_diag); transposes for B-operands.
__global__ void prep_weights(const float* __restrict__ Wp, const float* __restrict__ Wpd,
                             const float* __restrict__ Wzp, const float* __restrict__ plw,
                             const float* __restrict__ zlw, const float* __restrict__ f1w,
                             const float* __restrict__ f2w,
                             bf16* wWp, bf16* wWzp, bf16* wPT, bf16* wZT,
                             bf16* wF1T, bf16* wF2T) {
    int idx = blockIdx.x * blockDim.x + threadIdx.x;   // 262144
    int r = idx >> 9, c = idx & 511;
    int t = c * 512 + r;
    wWp[idx]  = (bf16)(Wp[idx] + (r == c ? Wpd[r] : 0.0f));
    wWzp[idx] = (bf16)Wzp[idx];
    wPT[idx]  = (bf16)plw[t];
    wZT[idx]  = (bf16)zlw[t];
    wF1T[idx] = (bf16)f1w[t];
    wF2T[idx] = (bf16)f2w[t];
}

// comb[b,h,i] = sum_c tp + sum_c tz + input[b,0,h,i]/5000  (deterministic, no atomics)
__global__ void combine_k(const bf16* __restrict__ tp, const bf16* __restrict__ tz,
                          const float* __restrict__ inp, bf16* __restrict__ comb) {
    long idx = (long)blockIdx.x * blockDim.x + threadIdx.x;  // 8388608
    int b  = (int)(idx >> 18);
    int hi = (int)(idx & 262143);
    long base = ((long)b << 21) + hi;            // b*C*H*I + hi (c=0)
    float s = inp[base] * (1.0f / 5000.0f);
#pragma unroll
    for (int c = 0; c < 8; ++c) {
        long o = base + ((long)c << 18);
        s += (float)tp[o] + (float)tz[o];
    }
    comb[idx] = (bf16)s;
}

extern "C" void kernel_launch(void* const* d_in, const int* in_sizes, int n_in,
                              void* d_out, int out_size, void* d_ws, size_t ws_size,
                              hipStream_t stream) {
    const float* inp   = (const float*)d_in[0];
    const float* pmask = (const float*)d_in[1];
    const float* Wp    = (const float*)d_in[2];
    const float* Wpd   = (const float*)d_in[3];
    const float* Wzp   = (const float*)d_in[4];
    const float* plw   = (const float*)d_in[5];
    const float* plb   = (const float*)d_in[6];
    const float* zlw   = (const float*)d_in[7];
    const float* zlb   = (const float*)d_in[8];
    const float* f1w   = (const float*)d_in[9];
    const float* f1b   = (const float*)d_in[10];
    const float* f2w   = (const float*)d_in[11];
    const float* f2b   = (const float*)d_in[12];

    char* ws = (char*)d_ws;
    const size_t XS_OFF = 0;                      // 32*8*512*512 bf16 = 128 MiB (tz reuses)
    const size_t TP_OFF = 134217728ull;           // 128 MiB
    const size_t CB_OFF = 268435456ull;           // comb: 16 MiB
    const size_t W_OFF  = 285212672ull;           // 6 x 512 KiB weights
    bf16* Xs   = (bf16*)(ws + XS_OFF);
    bf16* tp   = (bf16*)(ws + TP_OFF);
    bf16* comb = (bf16*)(ws + CB_OFF);
    bf16* wWp  = (bf16*)(ws + W_OFF);
    bf16* wWzp = (bf16*)(ws + W_OFF + 1 * 524288ull);
    bf16* wPT  = (bf16*)(ws + W_OFF + 2 * 524288ull);
    bf16* wZT  = (bf16*)(ws + W_OFF + 3 * 524288ull);
    bf16* wF1T = (bf16*)(ws + W_OFF + 4 * 524288ull);
    bf16* wF2T = (bf16*)(ws + W_OFF + 5 * 524288ull);

    prep_weights<<<1024, 256, 0, stream>>>(Wp, Wpd, Wzp, plw, zlw, f1w, f2w,
                                           wWp, wWzp, wPT, wZT, wF1T, wF2T);
    prep_x<<<65536, 256, 0, stream>>>(inp, pmask, Xs);

    // Stage A: tp = ht(ht(Wp_eff @ Xs) @ PT + p_lin_b), per (b,c)
    dim3 gbc(8, 256);
    fused_dual_gemm<1, 1, false, true><<<gbc, 256, 0, stream>>>(
        wWp, 0, Xs, 262144, wPT, nullptr, plb, tp, 262144);
    // Stage B: tz = ht(ht(Wzp @ tp) @ ZT + z_lin_b), per (b,c); tz overwrites Xs
    fused_dual_gemm<1, 1, false, true><<<gbc, 256, 0, stream>>>(
        wWzp, 0, tp, 262144, wZT, nullptr, zlb, Xs, 262144);
    // Reduce over C + residual
    combine_k<<<32768, 256, 0, stream>>>(tp, Xs, inp, comb);
    // Stage C: out = relu(comb @ F1T + f1b) @ F2T + f2b   (16384 rows)
    fused_dual_gemm<2, 0, true, false><<<dim3(256, 1), 256, 0, stream>>>(
        comb, 0, wF1T, 0, wF2T, f1b, f2b, d_out, 0);
}
